// LSTM_80616536146662
// MI455X (gfx1250) — compile-verified, matrix operand order checked by reading, not run on hardware
//
#include <hip/hip_runtime.h>
#include <hip/hip_bf16.h>

typedef __bf16 bf16_t;
typedef __attribute__((ext_vector_type(16))) __bf16 v16bf;
typedef __attribute__((ext_vector_type(8)))  __bf16 v8bf;
typedef __attribute__((ext_vector_type(8)))  float  v8f;

#define B_    64
#define T_    512
#define D_    512
#define H_    1024
#define G4    4096          // 4*H
#define MROWS (B_ * T_)     // 32768

// ---------------------------------------------------------------------------
// Fragment load: 16-bit A/B fragment for v_wmma_*_16x16x32.
// Per ISA 05_wmma layout: lanes 0-15 hold K = {k0..k0+7, k0+16..k0+23},
// lanes 16-31 hold the +8 variant. Rows are contiguous in K, so this is two
// 16-byte global/LDS loads.
// ---------------------------------------------------------------------------
__device__ __forceinline__ v16bf load_frag16(const bf16_t* __restrict__ rowptr,
                                             int k0, int kb) {
  v8bf lo = *(const v8bf*)(rowptr + k0 + kb);
  v8bf hi = *(const v8bf*)(rowptr + k0 + 16 + kb);
  return __builtin_shufflevector(lo, hi, 0, 1, 2, 3, 4, 5, 6, 7,
                                         8, 9, 10, 11, 12, 13, 14, 15);
}

// ---------------------------------------------------------------------------
// Init: zero the two grid-barrier counters (d_ws is poisoned each run).
// ---------------------------------------------------------------------------
__global__ void k_init(unsigned* __restrict__ cnt) {
  if (threadIdx.x < 2) cnt[threadIdx.x] = 0u;
}

// ---------------------------------------------------------------------------
// f32 -> bf16 cast (grid-stride).
// ---------------------------------------------------------------------------
__global__ void k_cast_bf16(const float* __restrict__ src,
                            bf16_t* __restrict__ dst, int n) {
  int i = blockIdx.x * blockDim.x + threadIdx.x;
  int stride = gridDim.x * blockDim.x;
  for (; i < n; i += stride) dst[i] = (bf16_t)src[i];
}

// ---------------------------------------------------------------------------
// W [K, 4096] f32 (row-major)  ->  WT [4096, K] bf16  (so B-fragments are
// contiguous 16B loads along K).
// ---------------------------------------------------------------------------
__global__ void k_transpose_cast(const float* __restrict__ W,
                                 bf16_t* __restrict__ WT, int K) {
  long i = blockIdx.x * (long)blockDim.x + threadIdx.x;
  long n = (long)K * G4;
  long stride = gridDim.x * (long)blockDim.x;
  for (; i < n; i += stride) {
    int nn = (int)(i / K);
    int kk = (int)(i - (long)nn * K);
    WT[i] = (bf16_t)W[(long)kk * G4 + nn];
  }
}

// ---------------------------------------------------------------------------
// Time-parallel projection: Z[M,4096] = X[M,K](bf16) * WT[4096,K](bf16) + bias
// Block = 256 threads = 8 waves; block tile = 128M x 64N; wave = 16M x 64N
// (4 accumulators, shared A-fragment). K stepped 32 per WMMA.
// ---------------------------------------------------------------------------
__global__ void __launch_bounds__(256)
k_gemm_xw(const bf16_t* __restrict__ X, const bf16_t* __restrict__ WT,
          const float* __restrict__ bias, float* __restrict__ Z, int K) {
  const int lane  = threadIdx.x & 31;
  const int w     = threadIdx.x >> 5;
  const int nbase = blockIdx.x * 64;
  const int mtile = blockIdx.y * 128 + w * 16;
  const int kb    = (lane >= 16) ? 8 : 0;
  const int nf    = lane & 15;

  v8f acc[4] = {};

  const bf16_t* arow = X + (long)(mtile + nf) * K;
  for (int k0 = 0; k0 < K; k0 += 32) {
    v16bf af = load_frag16(arow, k0, kb);
#pragma unroll
    for (int nt = 0; nt < 4; ++nt) {
      const bf16_t* brow = WT + (long)(nbase + nt * 16 + nf) * K;
      v16bf bfr = load_frag16(brow, k0, kb);
      acc[nt] = __builtin_amdgcn_wmma_f32_16x16x32_bf16(
          false, af, false, bfr, (short)0, acc[nt], false, false);
    }
  }

  const int mhalf = (lane >= 16) ? 8 : 0;
#pragma unroll
  for (int nt = 0; nt < 4; ++nt) {
    int n = nbase + nt * 16 + nf;
    float bv = bias[n];
#pragma unroll
    for (int r = 0; r < 8; ++r) {
      int m = mtile + r + mhalf;
      Z[(long)m * G4 + n] = acc[nt][r] + bv;
    }
  }
}

// ---------------------------------------------------------------------------
// Persistent recurrent kernel: 64 workgroups, one per 16 h-columns.
// Workgroup owns gate columns {j, j+H, j+2H, j+3H} for its 16 j's, all 64
// batch rows. 8 waves: wave w -> M-tile (w&3), gates {2*(w>>2), +1}.
// Per step: 2x32 WMMAs/wave against L2-resident WhT, gate fusion through a
// 16KB LDS z-buffer, c-state in registers, h ping-pong in global bf16,
// grid-wide atomic barrier per timestep.
// ---------------------------------------------------------------------------
__global__ void __launch_bounds__(256, 1)
k_lstm_rec(const float* __restrict__ xW,    // [B*T, 4096] f32 (bias included)
           const bf16_t* __restrict__ WhT,  // [4096, 1024] bf16
           float* __restrict__ outF,        // [B,T,H] f32 or nullptr
           bf16_t* __restrict__ outBf,      // [B*T,H] bf16 or nullptr
           bf16_t* __restrict__ hbufA, bf16_t* __restrict__ hbufB,
           unsigned* __restrict__ barrier_cnt) {
  __shared__ float zs[4 * 64 * 16];  // [gate][m][n], 16 KB

  const int tid   = threadIdx.x;
  const int lane  = tid & 31;
  const int w     = tid >> 5;
  const int hcol0 = blockIdx.x * 16;
  const int mtile = (w & 3) * 16;
  const int gbase = (w >> 2) * 2;
  const int kb    = (lane >= 16) ? 8 : 0;
  const int nf    = lane & 15;
  const int mhalf = (lane >= 16) ? 8 : 0;

  // element mapping for the fused gate/state phase: 4 elems per thread
  const int em  = tid >> 2;          // batch row 0..63
  const int en0 = (tid & 3) * 4;     // first of 4 consecutive n in 0..15
  float c_state[4] = {0.f, 0.f, 0.f, 0.f};

  const bf16_t* bg0 = WhT + (long)((gbase + 0) * H_ + hcol0 + nf) * H_;
  const bf16_t* bg1 = WhT + (long)((gbase + 1) * H_ + hcol0 + nf) * H_;

  bf16_t* hprev = hbufA;
  bf16_t* hcur  = hbufB;

  for (int t = 0; t < T_; ++t) {
    v8f acc0 = {};
    v8f acc1 = {};
    if (t > 0) {  // h_{-1} == 0: skip recurrent GEMM at t==0
      const bf16_t* arow = hprev + (mtile + nf) * H_;
#pragma unroll 4
      for (int k0 = 0; k0 < H_; k0 += 32) {
        v16bf af  = load_frag16(arow, k0, kb);
        v16bf bf0 = load_frag16(bg0, k0, kb);
        v16bf bf1 = load_frag16(bg1, k0, kb);
        acc0 = __builtin_amdgcn_wmma_f32_16x16x32_bf16(
            false, af, false, bf0, (short)0, acc0, false, false);
        acc1 = __builtin_amdgcn_wmma_f32_16x16x32_bf16(
            false, af, false, bf1, (short)0, acc1, false, false);
      }
    }
    // stage recurrent z into LDS (C/D layout: elem r -> m=r+8*(lane>=16), n=lane&15)
#pragma unroll
    for (int r = 0; r < 8; ++r) {
      int m = mtile + r + mhalf;
      zs[((gbase + 0) * 64 + m) * 16 + nf] = acc0[r];
      zs[((gbase + 1) * 64 + m) * 16 + nf] = acc1[r];
    }
    __syncthreads();

    // fused gates + state update
#pragma unroll
    for (int j = 0; j < 4; ++j) {
      int m = em, n = en0 + j;
      long xrow = ((long)m * T_ + t) * G4 + hcol0 + n;
      float zi = zs[(0 * 64 + m) * 16 + n] + xW[xrow + 0 * H_];
      float zf = zs[(1 * 64 + m) * 16 + n] + xW[xrow + 1 * H_];
      float zg = zs[(2 * 64 + m) * 16 + n] + xW[xrow + 2 * H_];
      float zo = zs[(3 * 64 + m) * 16 + n] + xW[xrow + 3 * H_];
      float ig = 1.0f / (1.0f + __expf(-zi));
      float fg = 1.0f / (1.0f + __expf(-zf));
      float gg = tanhf(zg);
      float og = 1.0f / (1.0f + __expf(-zo));
      float c  = fg * c_state[j] + ig * gg;
      c_state[j] = c;
      float h = og * tanhf(c);
      if (outF)  outF[((long)m * T_ + t) * H_ + hcol0 + n] = h;
      if (outBf) outBf[((long)m * T_ + t) * H_ + hcol0 + n] = (bf16_t)h;
      hcur[m * H_ + hcol0 + n] = (bf16_t)h;
    }
    __syncthreads();

    // grid-wide barrier: all h writes visible before next step's GEMM reads
    if (tid == 0) {
      __builtin_amdgcn_fence(__ATOMIC_RELEASE, "agent");
      atomicAdd(barrier_cnt, 1u);
      unsigned target = (unsigned)(t + 1) * gridDim.x;
      while (atomicAdd(barrier_cnt, 0u) < target) {
        __builtin_amdgcn_s_sleep(1);
      }
    }
    __syncthreads();
    __builtin_amdgcn_fence(__ATOMIC_ACQUIRE, "agent");

    bf16_t* tmp = hprev; hprev = hcur; hcur = tmp;
  }
}

// ---------------------------------------------------------------------------
// Host-side orchestration (graph-capture safe: only kernel launches on stream)
// ---------------------------------------------------------------------------
extern "C" void kernel_launch(void* const* d_in, const int* in_sizes, int n_in,
                              void* d_out, int out_size, void* d_ws, size_t ws_size,
                              hipStream_t stream) {
  (void)in_sizes; (void)n_in; (void)out_size; (void)ws_size;
  const float* x   = (const float*)d_in[0];
  const float* Wx0 = (const float*)d_in[1];
  const float* Wh0 = (const float*)d_in[2];
  const float* b0  = (const float*)d_in[3];
  const float* Wx1 = (const float*)d_in[4];
  const float* Wh1 = (const float*)d_in[5];
  const float* b1  = (const float*)d_in[6];
  float* out = (float*)d_out;

  char* ws = (char*)d_ws;
  size_t off = 0;
  auto take = [&](size_t bytes) -> char* {
    char* p = ws + off;
    off += (bytes + 255) & ~(size_t)255;
    return p;
  };
  float*  xW   = (float*) take((size_t)MROWS * G4 * sizeof(float));   // 512 MB
  bf16_t* xbf  = (bf16_t*)take((size_t)MROWS * D_ * sizeof(bf16_t));  // 32 MB
  bf16_t* h1bf = (bf16_t*)take((size_t)MROWS * H_ * sizeof(bf16_t));  // 64 MB
  bf16_t* WxT0 = (bf16_t*)take((size_t)G4 * D_ * sizeof(bf16_t));
  bf16_t* WhT0 = (bf16_t*)take((size_t)G4 * H_ * sizeof(bf16_t));
  bf16_t* WxT1 = (bf16_t*)take((size_t)G4 * H_ * sizeof(bf16_t));
  bf16_t* WhT1 = (bf16_t*)take((size_t)G4 * H_ * sizeof(bf16_t));
  bf16_t* hbA  = (bf16_t*)take((size_t)B_ * H_ * sizeof(bf16_t));
  bf16_t* hbB  = (bf16_t*)take((size_t)B_ * H_ * sizeof(bf16_t));
  unsigned* cnt = (unsigned*)take(256);

  k_init<<<1, 64, 0, stream>>>(cnt);
  k_cast_bf16<<<4096, 256, 0, stream>>>(x, xbf, MROWS * D_);
  k_transpose_cast<<<4096, 256, 0, stream>>>(Wx0, WxT0, D_);
  k_transpose_cast<<<4096, 256, 0, stream>>>(Wh0, WhT0, H_);
  k_transpose_cast<<<4096, 256, 0, stream>>>(Wx1, WxT1, H_);
  k_transpose_cast<<<4096, 256, 0, stream>>>(Wh1, WhT1, H_);

  dim3 gg(G4 / 64, MROWS / 128);

  // Layer 0
  k_gemm_xw<<<gg, 256, 0, stream>>>(xbf, WxT0, b0, xW, D_);
  k_lstm_rec<<<64, 256, 0, stream>>>(xW, WhT0, nullptr, h1bf, hbA, hbB, cnt + 0);

  // Layer 1
  k_gemm_xw<<<gg, 256, 0, stream>>>(h1bf, WxT1, b1, xW, H_);
  k_lstm_rec<<<64, 256, 0, stream>>>(xW, WhT1, out, nullptr, hbA, hbB, cnt + 1);
}
